// TSRNGistBlock_7834020348587
// MI455X (gfx1250) — matrix operand order repaired and checked by reading, not compiled, not run on hardware
//
#include <hip/hip_runtime.h>
#include <hip/hip_bf16.h>
#include <stdint.h>

// ---------------------------------------------------------------------------
// Problem dims (match setup_inputs): B=4, T=2048, d=2048, K=16, heads=16
// ---------------------------------------------------------------------------
#define DIM_B 4
#define DIM_T 2048
#define DIM_D 2048
#define DIM_K 16
#define DIM_H 16

typedef __attribute__((ext_vector_type(16))) __bf16 bf16x16;
typedef __attribute__((ext_vector_type(8)))  float  f32x8;

static __device__ __forceinline__ unsigned short f2bf(float f) {
    union { float f; unsigned u; } x; x.f = f;
    unsigned r = x.u + 0x7fffu + ((x.u >> 16) & 1u);   // round-to-nearest-even
    return (unsigned short)(r >> 16);
}

static __device__ __forceinline__ float block_reduce_sum_256(float v, float* red) {
    int tid = threadIdx.x;
    red[tid] = v; __syncthreads();
    #pragma unroll
    for (int s = 128; s > 0; s >>= 1) {
        if (tid < s) red[tid] += red[tid + s];
        __syncthreads();
    }
    float r = red[0]; __syncthreads();
    return r;
}

// wave32 shuffle reduction
static __device__ __forceinline__ float wave_reduce_sum(float v) {
    #pragma unroll
    for (int s = 16; s > 0; s >>= 1) v += __shfl_down(v, s, 32);
    return v;
}

// CDNA5 async global->LDS copy, 16B per lane (tracked by ASYNCcnt)
static __device__ __forceinline__ void async_copy_b128(unsigned lds_off, unsigned long long gaddr) {
    asm volatile("global_load_async_to_lds_b128 %0, %1, off"
                 :: "v"(lds_off), "v"(gaddr) : "memory");
}
static __device__ __forceinline__ void wait_async0() {
    asm volatile("s_wait_asynccnt 0x0" ::: "memory");
}

// ---------------------------------------------------------------------------
// f32 -> bf16 bulk convert
// ---------------------------------------------------------------------------
__global__ __launch_bounds__(256) void f32_to_bf16_kernel(
        const float* __restrict__ src, unsigned short* __restrict__ dst, size_t n) {
    size_t i = (size_t)blockIdx.x * blockDim.x + threadIdx.x;
    size_t stride = (size_t)gridDim.x * blockDim.x;
    for (; i < n; i += stride) dst[i] = f2bf(src[i]);
}

// ---------------------------------------------------------------------------
// cos/sin of sheaf thetas: [4, dh]
// ---------------------------------------------------------------------------
__global__ __launch_bounds__(256) void sheaf_trig_kernel(
        const float* __restrict__ th, float* __restrict__ c, float* __restrict__ s, int n) {
    int i = blockIdx.x * blockDim.x + threadIdx.x;
    if (i < n) { c[i] = __cosf(th[i]); s[i] = __sinf(th[i]); }
}

// ---------------------------------------------------------------------------
// Per-batch gist rotation angle -> cos/sin vectors [B, dh]
// ---------------------------------------------------------------------------
__global__ __launch_bounds__(256) void gist_theta_kernel(
        const float* __restrict__ gtheta,   // [B,K,dh]
        const float* __restrict__ gmag,     // [B,K,1]
        const float* __restrict__ gw,       // [B,K]
        const float* __restrict__ gstr,     // scalar
        float* __restrict__ cvec, float* __restrict__ svec, int K, int dh) {
    int b = blockIdx.x;
    __shared__ float w[DIM_K];
    __shared__ float wsum;
    if (threadIdx.x < K) w[threadIdx.x] = gw[b * K + threadIdx.x] * gmag[b * K + threadIdx.x];
    __syncthreads();
    if (threadIdx.x == 0) {
        float s = 0.f;
        for (int k = 0; k < K; ++k) s += w[k];
        wsum = s + 1e-8f;
    }
    __syncthreads();
    float sig = 1.f / (1.f + __expf(-gstr[0]));
    float inv = sig / wsum;
    for (int j = threadIdx.x; j < dh; j += blockDim.x) {
        float th = 0.f;
        #pragma unroll
        for (int k = 0; k < DIM_K; ++k) th += w[k] * gtheta[((size_t)b * K + k) * dh + j];
        th *= inv;
        cvec[(size_t)b * dh + j] = __cosf(th);
        svec[(size_t)b * dh + j] = __sinf(th);
    }
}

// ---------------------------------------------------------------------------
// gist_repr[row, o] = concat(theta,mag)[row,:] . gist_up_W[o,:] + b[o]
// Wave-per-output GEMV: lanes stride the feature dim (coalesced), shfl-reduce.
// grid = (d/8, rows); 8 waves per block.
// ---------------------------------------------------------------------------
__global__ __launch_bounds__(256) void gist_up_kernel(
        const float* __restrict__ gtheta, const float* __restrict__ gmag,
        const float* __restrict__ Wup,    // [d, dh+1]
        const float* __restrict__ bup,    // [d]
        float* __restrict__ repr, int dh, int d) {
    const int row  = blockIdx.y;
    const int wave = threadIdx.x >> 5;
    const int lane = threadIdx.x & 31;
    const int o    = blockIdx.x * 8 + wave;
    const int F    = dh + 1;
    const float* th = gtheta + (size_t)row * dh;
    const float* wr = Wup + (size_t)o * F;
    float acc = 0.f;
    for (int f = lane; f < dh; f += 32) acc += th[f] * wr[f];
    acc = wave_reduce_sum(acc);
    if (lane == 0)
        repr[(size_t)row * d + o] = acc + gmag[row] * wr[dh] + bup[o];
}

// out[row, o] = in[row,:] . W[o,:]   — wave-per-output GEMV, grid (d/8, rows)
__global__ __launch_bounds__(256) void proj_rows_kernel(
        const float* __restrict__ in, const float* __restrict__ W,
        float* __restrict__ out, int d, int Kin) {
    const int row  = blockIdx.y;
    const int wave = threadIdx.x >> 5;
    const int lane = threadIdx.x & 31;
    const int o    = blockIdx.x * 8 + wave;
    const float* ir = in + (size_t)row * Kin;
    const float* wr = W + (size_t)o * Kin;
    float acc = 0.f;
    for (int f = lane; f < Kin; f += 32) acc += ir[f] * wr[f];
    acc = wave_reduce_sum(acc);
    if (lane == 0) out[(size_t)row * d + o] = acc;
}

// ---------------------------------------------------------------------------
// Row LayerNorm: y = (x-mu)/sqrt(var+eps)*w + b   (one block per row)
// ---------------------------------------------------------------------------
__global__ __launch_bounds__(256) void ln_kernel(
        const float* __restrict__ x, const float* __restrict__ w, const float* __restrict__ b,
        float* __restrict__ y, int d) {
    __shared__ float red[256];
    size_t row = blockIdx.x;
    const float* xr = x + row * d;
    float s1 = 0.f, s2 = 0.f;
    for (int i = threadIdx.x; i < d; i += 256) { float v = xr[i]; s1 += v; s2 += v * v; }
    s1 = block_reduce_sum_256(s1, red);
    s2 = block_reduce_sum_256(s2, red);
    float mu = s1 / d;
    float inv = rsqrtf(s2 / d - mu * mu + 1e-5f);
    for (int i = threadIdx.x; i < d; i += 256)
        y[row * d + i] = (xr[i] - mu) * inv * w[i] + b[i];
}

// ---------------------------------------------------------------------------
// Sheaf pre-GEMM:  out = y - |alpha| * lap,  lap = 4*y - sum_idx rot_inv(y_nb)
// ---------------------------------------------------------------------------
__global__ __launch_bounds__(256) void sheaf_kernel(
        const float* __restrict__ y,
        const float* __restrict__ shc, const float* __restrict__ shs,   // [4,dh]
        const float* __restrict__ alpha,
        float* __restrict__ out_f, unsigned short* __restrict__ out_bf,
        int T, int d) {
    int row = blockIdx.x;              // b*T + t
    int t = row % T;
    int dh = d >> 1;
    const float* yt = y + (size_t)row * d;
    float a = fabsf(alpha[0]);
    for (int j = threadIdx.x; j < dh; j += 256) {
        float yr = yt[j], yi = yt[j + dh];
        float lr = 4.f * yr, li = 4.f * yi;
        #pragma unroll
        for (int idx = 0; idx < 4; ++idx) {
            int ad = 3 - idx;          // offsets -3,-2,-1,0
            float c = shc[idx * dh + j];
            float s = shs[idx * dh + j];
            float nr = 0.f, ni = 0.f;
            if (t - ad >= 0) {
                const float* yn = y + ((size_t)row - (size_t)ad) * d;
                nr = yn[j]; ni = yn[j + dh];
            }
            lr -= nr * c + ni * s;     // rot_inv real
            li -= -nr * s + ni * c;    // rot_inv imag
        }
        float orr = yr - a * lr;
        float oii = yi - a * li;
        size_t o0 = (size_t)row * d + j;
        out_f[o0] = orr; out_f[o0 + dh] = oii;
        out_bf[o0] = f2bf(orr); out_bf[o0 + dh] = f2bf(oii);
    }
}

// ---------------------------------------------------------------------------
// WMMA GEMM:  out[M,N] = A[M,K] * W[N,K]^T (+ up to 2 residual adds)
// bf16 in, f32 accumulate. Block tile 128x128, 8 waves (4x2), wave tile 32x64.
// Double-buffered LDS; async global->LDS staging overlaps WMMA of the live
// tile; B fragments via batched plain ds_load_b128 reads of n-major rows.
// ---------------------------------------------------------------------------
#define BM 128
#define BN 128
#define BK 32
#define LDSW 40   // 32 elems + 8 pad: row stride 80B, keeps 16B alignment

template <int NADD>
__global__ __launch_bounds__(256) void gemm_bf16_kernel(
        const unsigned short* __restrict__ A,   // [M,K] bf16 bits
        const unsigned short* __restrict__ W,   // [N,K] bf16 bits
        const float* __restrict__ add0,
        const float* __restrict__ add1,
        float* __restrict__ out,
        int M, int N, int K) {
    __shared__ __align__(16) unsigned short As[2][BM][LDSW];
    __shared__ __align__(16) unsigned short Bs[2][BN][LDSW];

    const int tid  = threadIdx.x;
    const int wave = tid >> 5;
    const int lane = tid & 31;
    const int half = lane >> 4;
    const int l16  = lane & 15;
    const int wm   = wave >> 1;          // 0..3 -> 32-row strip
    const int wn   = wave & 1;           // 0..1 -> 64-col strip

    const int n0 = blockIdx.x * BN;
    const int m0 = blockIdx.y * BM;

    // staging coords: each thread copies 32B of one A row + 32B of one W row
    const int lrow  = tid >> 1;          // 0..127
    const int lhalf = tid & 1;

    const unsigned asb[2] = { (unsigned)(uintptr_t)&As[0][0][0],
                              (unsigned)(uintptr_t)&As[1][0][0] };
    const unsigned bsb[2] = { (unsigned)(uintptr_t)&Bs[0][0][0],
                              (unsigned)(uintptr_t)&Bs[1][0][0] };
    const unsigned stg = (unsigned)(lrow * (LDSW * 2) + lhalf * 32);
    const unsigned long long gA = (unsigned long long)(A + (size_t)(m0 + lrow) * K + lhalf * 16);
    const unsigned long long gB = (unsigned long long)(W + (size_t)(n0 + lrow) * K + lhalf * 16);

    auto stage = [&](int buf, int kk) {
        unsigned long long koff = (unsigned long long)kk * 2;   // bytes
        async_copy_b128(asb[buf] + stg,      gA + koff);
        async_copy_b128(asb[buf] + stg + 16, gA + koff + 16);
        async_copy_b128(bsb[buf] + stg,      gB + koff);
        async_copy_b128(bsb[buf] + stg + 16, gB + koff + 16);
    };

    f32x8 acc[2][4];
    #pragma unroll
    for (int mi = 0; mi < 2; ++mi)
        #pragma unroll
        for (int j = 0; j < 4; ++j) acc[mi][j] = (f32x8)0.f;

    // prologue: stage first tile
    stage(0, 0);
    wait_async0();
    __syncthreads();

    for (int kk = 0, it = 0; kk < K; kk += BK, ++it) {
        const int rb = it & 1;
        if (kk + BK < K) stage(rb ^ 1, kk + BK);   // DMA next tile under compute

        // ---- A fragments: rows wm*32 + {0,16} + l16; K-in-lane per A layout ----
        union { bf16x16 v; uint4 u[2]; } af[2];
        #pragma unroll
        for (int mi = 0; mi < 2; ++mi) {
            const int mloc = wm * 32 + mi * 16 + l16;
            af[mi].u[0] = *(const uint4*)&As[rb][mloc][half * 8];
            af[mi].u[1] = *(const uint4*)&As[rb][mloc][16 + half * 8];
        }

        // ---- all 4 B fragments in one batch of ds_load_b128 loads ----
        union { bf16x16 v; uint4 u[2]; } bfr[4];
        {
            unsigned a0 = bsb[rb] + (unsigned)((wn * 64 + 0 * 16 + l16) * (LDSW * 2)) + (unsigned)(half * 16);
            unsigned a1 = bsb[rb] + (unsigned)((wn * 64 + 1 * 16 + l16) * (LDSW * 2)) + (unsigned)(half * 16);
            unsigned a2 = bsb[rb] + (unsigned)((wn * 64 + 2 * 16 + l16) * (LDSW * 2)) + (unsigned)(half * 16);
            unsigned a3 = bsb[rb] + (unsigned)((wn * 64 + 3 * 16 + l16) * (LDSW * 2)) + (unsigned)(half * 16);
            asm volatile(
                "ds_load_b128 %0, %8\n\t"
                "ds_load_b128 %1, %9\n\t"
                "ds_load_b128 %2, %10\n\t"
                "ds_load_b128 %3, %11\n\t"
                "ds_load_b128 %4, %12\n\t"
                "ds_load_b128 %5, %13\n\t"
                "ds_load_b128 %6, %14\n\t"
                "ds_load_b128 %7, %15\n\t"
                "s_wait_dscnt 0x0"
                : "=&v"(bfr[0].u[0]), "=&v"(bfr[0].u[1]),
                  "=&v"(bfr[1].u[0]), "=&v"(bfr[1].u[1]),
                  "=&v"(bfr[2].u[0]), "=&v"(bfr[2].u[1]),
                  "=&v"(bfr[3].u[0]), "=&v"(bfr[3].u[1])
                : "v"(a0), "v"(a0 + 32),
                  "v"(a1), "v"(a1 + 32),
                  "v"(a2), "v"(a2 + 32),
                  "v"(a3), "v"(a3 + 32)
                : "memory");
        }

        // ---- 8 back-to-back WMMAs ----
        #pragma unroll
        for (int j = 0; j < 4; ++j)
            #pragma unroll
            for (int mi = 0; mi < 2; ++mi)
                acc[mi][j] = __builtin_amdgcn_wmma_f32_16x16x32_bf16(
                                 false, af[mi].v, false, bfr[j].v, (short)0,
                                 acc[mi][j], false, false);

        wait_async0();
        __syncthreads();
    }

    // ---- epilogue: D rows are (base + v) contiguous since row = v + half*8 ----
    #pragma unroll
    for (int mi = 0; mi < 2; ++mi) {
        const int rbase = m0 + wm * 32 + mi * 16 + half * 8;
        #pragma unroll
        for (int j = 0; j < 4; ++j) {
            const int c = n0 + wn * 64 + j * 16 + l16;
            size_t p = (size_t)rbase * N + c;
            #pragma unroll
            for (int v = 0; v < 8; ++v) {
                float val = acc[mi][j][v];
                if (NADD >= 1) val += add0[p];
                if (NADD >= 2) val += add1[p];
                out[p] = val;
                p += (size_t)N;
            }
        }
    }
}

// ---------------------------------------------------------------------------
// Fused gist rotation: xn=LN(x1); rot=R(xn); gr=LN(rot); x2=x1+gr-xn; xg=LN(x2)
// ---------------------------------------------------------------------------
__global__ __launch_bounds__(256) void gist_fuse_kernel(
        const float* __restrict__ x1,
        const float* __restrict__ cvec, const float* __restrict__ svec,  // [B,dh]
        const float* __restrict__ w1, const float* __restrict__ b1,      // ln_gist
        const float* __restrict__ w2, const float* __restrict__ b2,      // gr_ln
        const float* __restrict__ w3, const float* __restrict__ b3,      // ln_gca
        float* __restrict__ x2out, unsigned short* __restrict__ xgbf,
        int T, int d) {
    __shared__ float sx[DIM_D];
    __shared__ float sxn[DIM_D];
    __shared__ float sr[DIM_D];
    __shared__ float red[256];

    int row = blockIdx.x;
    int b = row / T;
    int dh = d >> 1;
    const float* xr = x1 + (size_t)row * d;

    float s1 = 0.f, s2 = 0.f;
    for (int i = threadIdx.x; i < d; i += 256) { float v = xr[i]; sx[i] = v; s1 += v; s2 += v * v; }
    s1 = block_reduce_sum_256(s1, red);
    s2 = block_reduce_sum_256(s2, red);
    float mu = s1 / d, inv = rsqrtf(s2 / d - mu * mu + 1e-5f);

    for (int i = threadIdx.x; i < d; i += 256) sxn[i] = (sx[i] - mu) * inv * w1[i] + b1[i];
    __syncthreads();

    s1 = 0.f; s2 = 0.f;
    for (int j = threadIdx.x; j < dh; j += 256) {
        float r = sxn[j], im = sxn[j + dh];
        float c = cvec[(size_t)b * dh + j], s = svec[(size_t)b * dh + j];
        float rr = r * c - im * s;
        float ri = r * s + im * c;
        sr[j] = rr; sr[j + dh] = ri;
        s1 += rr + ri; s2 += rr * rr + ri * ri;
    }
    s1 = block_reduce_sum_256(s1, red);
    s2 = block_reduce_sum_256(s2, red);
    float mu2 = s1 / d, inv2 = rsqrtf(s2 / d - mu2 * mu2 + 1e-5f);

    s1 = 0.f; s2 = 0.f;
    for (int i = threadIdx.x; i < d; i += 256) {
        float gr = (sr[i] - mu2) * inv2 * w2[i] + b2[i];
        float v = sx[i] + gr - sxn[i];
        sx[i] = v;
        x2out[(size_t)row * d + i] = v;
        s1 += v; s2 += v * v;
    }
    s1 = block_reduce_sum_256(s1, red);
    s2 = block_reduce_sum_256(s2, red);
    float mu3 = s1 / d, inv3 = rsqrtf(s2 / d - mu3 * mu3 + 1e-5f);

    for (int i = threadIdx.x; i < d; i += 256)
        xgbf[(size_t)row * d + i] = f2bf((sx[i] - mu3) * inv3 * w3[i] + b3[i]);
}

// ---------------------------------------------------------------------------
// Cross attention over K=16 gists, 16 heads of 128. One block per (b,t) row.
// ---------------------------------------------------------------------------
__global__ __launch_bounds__(256) void attn_kernel(
        const float* __restrict__ Q,    // [B*T, d]
        const float* __restrict__ Kh,   // [B*K, d]
        const float* __restrict__ V,    // [B*K, d]
        unsigned short* __restrict__ ctx_bf,
        int T, int d, int Kg, int H) {
    __shared__ float sq[DIM_D];
    __shared__ float sred[256];
    __shared__ float sp[DIM_K];

    int row = blockIdx.x;
    int b = row / T;
    int hd = d / H;                     // 128
    int tid = threadIdx.x;
    int kk = tid >> 4, u = tid & 15;    // 16 k-groups x 16 threads
    float scale = rsqrtf((float)hd);

    const float* qr = Q + (size_t)row * d;
    for (int i = tid; i < d; i += 256) sq[i] = qr[i];
    __syncthreads();

    for (int h = 0; h < H; ++h) {
        float p = 0.f;
        const float* kr = Kh + ((size_t)(b * Kg + kk)) * d + h * hd;
        int ePer = hd / 16;             // 8
        for (int e = u * ePer; e < u * ePer + ePer; ++e) p += sq[h * hd + e] * kr[e];
        sred[tid] = p; __syncthreads();
        #pragma unroll
        for (int s = 8; s > 0; s >>= 1) {
            if (u < s) sred[tid] += sred[tid + s];
            __syncthreads();
        }
        if (u == 0) sp[kk] = sred[tid] * scale;
        __syncthreads();
        if (tid == 0) {
            float mx = -1e30f;
            for (int k = 0; k < Kg; ++k) mx = fmaxf(mx, sp[k]);
            float sum = 0.f;
            for (int k = 0; k < Kg; ++k) { float e = __expf(sp[k] - mx); sp[k] = e; sum += e; }
            float is = 1.f / sum;
            for (int k = 0; k < Kg; ++k) sp[k] *= is;
        }
        __syncthreads();
        if (tid < hd) {
            float acc = 0.f;
            for (int k = 0; k < Kg; ++k)
                acc += sp[k] * V[((size_t)(b * Kg + k)) * d + h * hd + tid];
            ctx_bf[(size_t)row * d + h * hd + tid] = f2bf(acc);
        }
        __syncthreads();
    }
}

// ---------------------------------------------------------------------------
// Launch
// ---------------------------------------------------------------------------
extern "C" void kernel_launch(void* const* d_in, const int* in_sizes, int n_in,
                              void* d_out, int out_size, void* d_ws, size_t ws_size,
                              hipStream_t stream) {
    const float* x      = (const float*)d_in[0];
    const float* gtheta = (const float*)d_in[1];
    const float* gmag   = (const float*)d_in[2];
    const float* gw     = (const float*)d_in[3];
    const float* lnsw   = (const float*)d_in[4];
    const float* lnsb   = (const float*)d_in[5];
    const float* sheafT = (const float*)d_in[6];
    const float* alpha  = (const float*)d_in[7];
    const float* corrW  = (const float*)d_in[8];
    const float* lngw   = (const float*)d_in[9];
    const float* lngb   = (const float*)d_in[10];
    const float* gstr   = (const float*)d_in[11];
    const float* grw    = (const float*)d_in[12];
    const float* grb    = (const float*)d_in[13];
    const float* lncw   = (const float*)d_in[14];
    const float* lncb   = (const float*)d_in[15];
    const float* Wq     = (const float*)d_in[16];
    const float* Wk     = (const float*)d_in[17];
    const float* Wv     = (const float*)d_in[18];
    const float* Wo     = (const float*)d_in[19];
    const float* WupW   = (const float*)d_in[20];
    const float* Wupb   = (const float*)d_in[21];

    const int B = DIM_B, T = DIM_T, d = DIM_D, K = DIM_K, H = DIM_H;
    const int BT = B * T, dh = d / 2;
    const size_t nBTd = (size_t)BT * d;
    const size_t ndd  = (size_t)d * d;

    // ---- workspace carve (all offsets 256B aligned) ----
    char* w = (char*)d_ws;
    size_t off = 0;
    auto carve = [&](size_t bytes) { void* p = w + off; off += (bytes + 255) & ~(size_t)255; return p; };
    float*          R0    = (float*)carve(nBTd * 4);          // y (LN sheaf) -> x2
    float*          R1    = (float*)carve(nBTd * 4);          // s_out f32 -> Q
    float*          R2    = (float*)carve(nBTd * 4);          // x1
    unsigned short* R3    = (unsigned short*)carve(nBTd * 2); // bf16 A staging
    unsigned short* WcBF  = (unsigned short*)carve(ndd * 2);
    unsigned short* WqBF  = (unsigned short*)carve(ndd * 2);
    unsigned short* WoBF  = (unsigned short*)carve(ndd * 2);
    float*          cvec  = (float*)carve((size_t)B * dh * 4);
    float*          svec  = (float*)carve((size_t)B * dh * 4);
    float*          shc   = (float*)carve((size_t)4 * dh * 4);
    float*          shs   = (float*)carve((size_t)4 * dh * 4);
    float*          repr  = (float*)carve((size_t)B * K * d * 4);
    float*          KhP   = (float*)carve((size_t)B * K * d * 4);
    float*          VP    = (float*)carve((size_t)B * K * d * 4);
    (void)ws_size; (void)n_in; (void)in_sizes; (void)out_size;

    // 1) weight conversions + small precomputes
    f32_to_bf16_kernel<<<2048, 256, 0, stream>>>(corrW, WcBF, ndd);
    f32_to_bf16_kernel<<<2048, 256, 0, stream>>>(Wq,    WqBF, ndd);
    f32_to_bf16_kernel<<<2048, 256, 0, stream>>>(Wo,    WoBF, ndd);
    sheaf_trig_kernel<<<(4 * dh + 255) / 256, 256, 0, stream>>>(sheafT, shc, shs, 4 * dh);
    gist_theta_kernel<<<B, 256, 0, stream>>>(gtheta, gmag, gw, gstr, cvec, svec, K, dh);
    {
        dim3 g(d / 8, B * K);
        gist_up_kernel<<<g, 256, 0, stream>>>(gtheta, gmag, WupW, Wupb, repr, dh, d);
        proj_rows_kernel<<<g, 256, 0, stream>>>(repr, Wk, KhP, d, d);
        proj_rows_kernel<<<g, 256, 0, stream>>>(repr, Wv, VP,  d, d);
    }

    // 2) sheaf branch
    ln_kernel<<<BT, 256, 0, stream>>>(x, lnsw, lnsb, R0, d);
    sheaf_kernel<<<BT, 256, 0, stream>>>(R0, shc, shs, alpha, R1, R3, T, d);

    // 3) GEMM1: x1 = x + s_out + s_out @ corr_W^T
    dim3 ggrid(d / BN, BT / BM);
    gemm_bf16_kernel<2><<<ggrid, 256, 0, stream>>>(R3, WcBF, x, R1, R2, BT, d, d);

    // 4) gist rotation residual (+ produce xg bf16 for Q GEMM)
    gist_fuse_kernel<<<BT, 256, 0, stream>>>(R2, cvec, svec, lngw, lngb, grw, grb,
                                             lncw, lncb, R0, R3, T, d);

    // 5) GEMM2: Q = xg @ Wq^T
    gemm_bf16_kernel<0><<<ggrid, 256, 0, stream>>>(R3, WqBF, nullptr, nullptr, R1, BT, d, d);

    // 6) attention over gists -> ctx bf16
    attn_kernel<<<BT, 256, 0, stream>>>(R1, KhP, VP, R3, T, d, K, H);

    // 7) GEMM3: out = x2 + ctx @ Wo^T
    gemm_bf16_kernel<1><<<ggrid, 256, 0, stream>>>(R3, WoBF, R0, nullptr, (float*)d_out, BT, d, d);
}
